// RelationalPointerNet_9500467659502
// MI455X (gfx1250) — compile-verified, matrix operand order checked by reading, not run on hardware
//
#include <hip/hip_runtime.h>
#include <hip/hip_bf16.h>

// Problem constants (from reference)
#define BB 4
#define SS 512
#define HH 256
#define NREL 99
#define NRELP 128           // NREL padded to multiple of 32 for 32x32 macro-tiles

typedef __bf16 bf16;
typedef __attribute__((ext_vector_type(16))) __bf16 v16bf;
typedef __attribute__((ext_vector_type(8)))  float  v8f;

// ---------------------------------------------------------------------------
// f32 -> bf16 conversion helpers (one-time operand preparation)
// ---------------------------------------------------------------------------
__global__ void cvt_f32_to_bf16(const float* __restrict__ src,
                                bf16* __restrict__ dst, int n) {
    int i = blockIdx.x * blockDim.x + threadIdx.x;
    if (i < n) dst[i] = (bf16)src[i];
}

// Convert with zero padding (for rel_emb rows 99..127)
__global__ void cvt_f32_to_bf16_pad(const float* __restrict__ src,
                                    bf16* __restrict__ dst, int n_src, int n_total) {
    int i = blockIdx.x * blockDim.x + threadIdx.x;
    if (i < n_total) dst[i] = (i < n_src) ? (bf16)src[i] : (bf16)0.0f;
}

// ---------------------------------------------------------------------------
// WMMA fragment load, 16-bit elements, row-major source with leading dim ld.
// ISA 7.12.2 (16-bit A 16x32): lanes 0-15 hold row r0+lane, K = k0+0..7 (v16bf
// elems 0..7) and k0+16..23 (elems 8..15); lanes 16-31 hold K = k0+8..15 and
// k0+24..31.  The B operand of C = A * W^T has the identical addressing
// pattern on W (lane <-> output column, contiguous along K).
// ---------------------------------------------------------------------------
__device__ __forceinline__ v16bf load_frag_rm(const bf16* __restrict__ base,
                                              int ld, int r0, int k0) {
    int lane = threadIdx.x & 31;
    int r  = r0 + (lane & 15);
    int kb = k0 + ((lane & 16) ? 8 : 0);
    const bf16* p = base + (size_t)r * ld + kb;
    v16bf f;
#pragma unroll
    for (int i = 0; i < 8; ++i) {
        f[i]     = p[i];        // K = kb .. kb+7      (16B contiguous)
        f[i + 8] = p[i + 16];   // K = kb+16 .. kb+23  (16B contiguous)
    }
    return f;
}

__device__ __forceinline__ v8f wmma_bf16(const v16bf& a, const v16bf& b, const v8f& c) {
    return __builtin_amdgcn_wmma_f32_16x16x32_bf16(
        /*neg_a=*/false, a, /*neg_b=*/false, b,
        /*c_mod=*/(short)0, c, /*reuse_a=*/false, /*reuse_b=*/false);
}

// Store one 16x16 f32 sub-tile (C/D layout ISA 7.12.2):
//   lane<16 -> N=n0+lane, M=m0+i ; lane>=16 -> N=n0+lane-16, M=m0+8+i
__device__ __forceinline__ void store_tile(const v8f& c,
                                           float* __restrict__ Cf,
                                           bf16*  __restrict__ Cb,
                                           const float* __restrict__ bias,
                                           size_t cOff, int N, int m0, int n0) {
    const int lane = threadIdx.x & 31;
    const int n  = n0 + (lane & 15);
    const int mb = m0 + ((lane & 16) ? 8 : 0);
    const float bv = bias ? bias[n] : 0.0f;
#pragma unroll
    for (int i = 0; i < 8; ++i) {
        float v = c[i] + bv;
        if (Cf) Cf[cOff + (size_t)(mb + i) * N + n] = v;
        if (Cb) Cb[cOff + (size_t)(mb + i) * N + n] = (bf16)v;
    }
}

// ---------------------------------------------------------------------------
// Register-blocked tile GEMM:  C[M][N] = A[M][K] * W[N][K]^T (+ bias[n])
// Each wave owns a 32x32 macro-tile: 2 A frags x 2 B frags -> 4 WMMAs per
// 32-deep k-step (1 fragment load per WMMA instead of 2).  8 waves
// (256 threads, wave32) per block.  M, N must be multiples of 32; K of 32.
// blockIdx.z = batch; strides in elements.  Cf: f32 out, Cb: bf16 out.
// ---------------------------------------------------------------------------
__global__ void wmma_gemm_abT(const bf16* __restrict__ A,
                              const bf16* __restrict__ W,
                              const float* __restrict__ bias,
                              float* __restrict__ Cf,
                              bf16*  __restrict__ Cb,
                              int M, int N, int K,
                              long strideA, long strideW, long strideC) {
    const int batch = blockIdx.z;
    A += (size_t)batch * strideA;
    W += (size_t)batch * strideW;
    const size_t cOff = (size_t)batch * strideC;

    const int ntilesN = N >> 5;                       // 32-wide macro columns
    const int ntiles  = (M >> 5) * ntilesN;
    const int tile    = blockIdx.x * (blockDim.x >> 5) + (threadIdx.x >> 5);
    if (tile >= ntiles) return;                       // wave-uniform exit: EXEC all-1s
    const int m0 = (tile / ntilesN) << 5;
    const int n0 = (tile % ntilesN) << 5;

    v8f c00 = {}, c01 = {}, c10 = {}, c11 = {};
#pragma unroll 2
    for (int k0 = 0; k0 < K; k0 += 32) {
        v16bf a0 = load_frag_rm(A, K, m0,      k0);
        v16bf a1 = load_frag_rm(A, K, m0 + 16, k0);
        v16bf b0 = load_frag_rm(W, K, n0,      k0);
        v16bf b1 = load_frag_rm(W, K, n0 + 16, k0);
        c00 = wmma_bf16(a0, b0, c00);
        c01 = wmma_bf16(a0, b1, c01);
        c10 = wmma_bf16(a1, b0, c10);
        c11 = wmma_bf16(a1, b1, c11);
    }

    store_tile(c00, Cf, Cb, bias, cOff, N, m0,      n0);
    store_tile(c01, Cf, Cb, bias, cOff, N, m0,      n0 + 16);
    store_tile(c10, Cf, Cb, bias, cOff, N, m0 + 16, n0);
    store_tile(c11, Cf, Cb, bias, cOff, N, m0 + 16, n0 + 16);
}

// ---------------------------------------------------------------------------
// Fused logits + softmax:
//   logits[row,k] = (qk[row,k] + qr_table[row, relations[row,k]]) * 1/sqrt(H)
//   out[row,:]    = softmax(logits[row,:])
// One block (256 threads) per row of length S=512; 2 elements per thread.
// ---------------------------------------------------------------------------
__global__ void logits_softmax(const float* __restrict__ qk,
                               const float* __restrict__ qr_table,
                               const int*   __restrict__ relations,
                               float* __restrict__ out) {
    const int row = blockIdx.x;                 // 0 .. B*S-1
    const int tid = threadIdx.x;
    const size_t rbase = (size_t)row * SS;
    const size_t tbase = (size_t)row * NRELP;
    const float scale = 0.0625f;                // 1/sqrt(256)

    const int k0 = tid, k1 = tid + 256;
    const int r0 = relations[rbase + k0];
    const int r1 = relations[rbase + k1];
    float l0 = (qk[rbase + k0] + qr_table[tbase + r0]) * scale;
    float l1 = (qk[rbase + k1] + qr_table[tbase + r1]) * scale;

    __shared__ float red[8];
    const int wave = tid >> 5;
    const int lane = tid & 31;

    // ---- block max ----
    float m = fmaxf(l0, l1);
#pragma unroll
    for (int off = 16; off > 0; off >>= 1)
        m = fmaxf(m, __shfl_xor(m, off, 32));
    if (lane == 0) red[wave] = m;
    __syncthreads();
    float bm = red[0];
#pragma unroll
    for (int w = 1; w < 8; ++w) bm = fmaxf(bm, red[w]);
    __syncthreads();

    // ---- exp + block sum ----
    float e0 = __expf(l0 - bm);
    float e1 = __expf(l1 - bm);
    float s = e0 + e1;
#pragma unroll
    for (int off = 16; off > 0; off >>= 1)
        s += __shfl_xor(s, off, 32);
    if (lane == 0) red[wave] = s;
    __syncthreads();
    float bs = 0.0f;
#pragma unroll
    for (int w = 0; w < 8; ++w) bs += red[w];
    const float inv = 1.0f / bs;

    out[rbase + k0] = e0 * inv;
    out[rbase + k1] = e1 * inv;
}

// ---------------------------------------------------------------------------
// Launch
// Inputs (setup_inputs order):
//   0 queries [B,S,H] f32   1 keys [B,S,H] f32   2 relations [B,S,S] int
//   3 Wq [H,H] f32          4 bq [H] f32
//   5 Wk [H,H] f32          6 bk [H] f32         7 rel_emb [NREL,H] f32
// Output: softmax probs [B,S,S] f32
// ---------------------------------------------------------------------------
extern "C" void kernel_launch(void* const* d_in, const int* in_sizes, int n_in,
                              void* d_out, int out_size, void* d_ws, size_t ws_size,
                              hipStream_t stream) {
    const float* queries  = (const float*)d_in[0];
    const float* keys     = (const float*)d_in[1];
    const int*   relations= (const int*)  d_in[2];
    const float* Wq       = (const float*)d_in[3];
    const float* bq       = (const float*)d_in[4];
    const float* Wk       = (const float*)d_in[5];
    const float* bk       = (const float*)d_in[6];
    const float* rel_emb  = (const float*)d_in[7];
    float* out            = (float*)d_out;

    const int MQ = BB * SS;        // 2048 rows of q/k

    // ---- workspace carve-up (bytes, 256-aligned) ----
    char* ws = (char*)d_ws;
    size_t off = 0;
    auto alloc = [&](size_t bytes) { char* p = ws + off; off = (off + bytes + 255) & ~(size_t)255; return p; };
    bf16* q_in_bf = (bf16*)alloc((size_t)MQ * HH * 2);       // queries bf16
    bf16* k_in_bf = (bf16*)alloc((size_t)MQ * HH * 2);       // keys bf16
    bf16* Wq_bf   = (bf16*)alloc((size_t)HH * HH * 2);
    bf16* Wk_bf   = (bf16*)alloc((size_t)HH * HH * 2);
    bf16* rel_bf  = (bf16*)alloc((size_t)NRELP * HH * 2);    // zero-padded rows
    bf16* q_bf    = (bf16*)alloc((size_t)MQ * HH * 2);       // projected q
    bf16* k_bf    = (bf16*)alloc((size_t)MQ * HH * 2);       // projected k
    float* qk     = (float*)alloc((size_t)BB * SS * SS * 4); // q @ k^T
    float* qr     = (float*)alloc((size_t)MQ * NRELP * 4);   // q @ rel_emb^T
    (void)ws_size;

    // ---- 1) convert operands to bf16 ----
    {
        int n = MQ * HH;
        cvt_f32_to_bf16<<<dim3((n + 255) / 256), dim3(256), 0, stream>>>(queries, q_in_bf, n);
        cvt_f32_to_bf16<<<dim3((n + 255) / 256), dim3(256), 0, stream>>>(keys,    k_in_bf, n);
        int nw = HH * HH;
        cvt_f32_to_bf16<<<dim3((nw + 255) / 256), dim3(256), 0, stream>>>(Wq, Wq_bf, nw);
        cvt_f32_to_bf16<<<dim3((nw + 255) / 256), dim3(256), 0, stream>>>(Wk, Wk_bf, nw);
        int nr_src = NREL * HH, nr_tot = NRELP * HH;
        cvt_f32_to_bf16_pad<<<dim3((nr_tot + 255) / 256), dim3(256), 0, stream>>>(rel_emb, rel_bf, nr_src, nr_tot);
    }

    // ---- 2) projections: q = queries*Wq^T + bq ; k = keys*Wk^T + bk  (bf16 out)
    {
        int ntiles = (MQ / 32) * (HH / 32);              // 512 macro-tiles
        dim3 grid((ntiles + 7) / 8, 1, 1), block(256);
        wmma_gemm_abT<<<grid, block, 0, stream>>>(q_in_bf, Wq_bf, bq, nullptr, q_bf,
                                                  MQ, HH, HH, 0, 0, 0);
        wmma_gemm_abT<<<grid, block, 0, stream>>>(k_in_bf, Wk_bf, bk, nullptr, k_bf,
                                                  MQ, HH, HH, 0, 0, 0);
    }

    // ---- 3) qk = q @ k^T per batch (f32 out) ----
    {
        int ntiles = (SS / 32) * (SS / 32);              // 256 macro-tiles per batch
        dim3 grid((ntiles + 7) / 8, 1, BB), block(256);
        wmma_gemm_abT<<<grid, block, 0, stream>>>(q_bf, k_bf, nullptr, qk, nullptr,
                                                  SS, SS, HH,
                                                  (long)SS * HH, (long)SS * HH, (long)SS * SS);
    }

    // ---- 4) qr_table = q @ rel_emb^T  [MQ x NRELP] (f32 out) ----
    {
        int ntiles = (MQ / 32) * (NRELP / 32);           // 256 macro-tiles
        dim3 grid((ntiles + 7) / 8, 1, 1), block(256);
        wmma_gemm_abT<<<grid, block, 0, stream>>>(q_bf, rel_bf, nullptr, qr, nullptr,
                                                  MQ, NRELP, HH, 0, 0, 0);
    }

    // ---- 5) fused relation-lookup + scale + softmax ----
    {
        dim3 grid(BB * SS), block(256);
        logits_softmax<<<grid, block, 0, stream>>>(qk, qr, relations, out);
    }
}
